// Attention_14877766714195
// MI455X (gfx1250) — matrix operand order repaired and checked by reading, not compiled, not run on hardware
//
#include <hip/hip_runtime.h>
#include <hip/hip_bf16.h>

// ---------------------------------------------------------------------------
// Multi-head attention forward for MI455X (gfx1250), wave32 + WMMA bf16.
//   B=2, S=2048, SIZE=1024, H=16, D=64.
// Pipeline:
//   1) Qb = (q @ Wq^T + bq) * 0.125   -> bf16 ws   (scale 1/sqrt(64) folded in)
//      Kb = (k @ Wk^T + bk)           -> bf16 ws
//      Vb = (v @ Wv^T + bv)           -> bf16 ws
//   2) flash attention per (b,h,qtile) -> Ctx bf16 ws ([B*S, 1024] merged heads)
//   3) out = Ctx @ Wo^T + bo          -> f32 d_out
// Matmuls: v_wmma_f32_16x16x32_bf16 (fp32 accumulate).
// bf16 tiles that need no conversion are staged with
// global_load_async_to_lds_b128 (ASYNCcnt) instead of load+ds_store.
// Workspace: 4 x (4096*1024) bf16 = 32 MB.
// ---------------------------------------------------------------------------

typedef __attribute__((ext_vector_type(16))) __bf16 v16bf;
typedef __attribute__((ext_vector_type(8)))  __bf16 v8bf;
typedef __attribute__((ext_vector_type(8)))  float  v8f;

#define ATTN_B   2
#define ATTN_S   2048
#define ATTN_SZ  1024
#define ATTN_H   16
#define ATTN_D   64

union FragU { v16bf v; v8bf h[2]; };

// A-matrix 16x32 bf16 fragment (ISA 7.12.2): lane (half=lane/16, m=lane%16)
// holds k = {8*half..8*half+7} and {16+8*half..16+8*half+7} of row m.
__device__ __forceinline__ v16bf load_a_frag(const __bf16* base, int half) {
    FragU u;
    u.h[0] = *(const v8bf*)(base + 8 * half);
    u.h[1] = *(const v8bf*)(base + 16 + 8 * half);
    return u.v;
}

// B-matrix 32x16 bf16 fragment: lane n=lane%16, kh=lane/16 holds
// k = {8*kh..8*kh+7} and {16+8*kh..16+8*kh+7} of column n (we feed B^T rows).
__device__ __forceinline__ v16bf load_b_frag(const __bf16* base, int half) {
    FragU u;
    u.h[0] = *(const v8bf*)(base + 8 * half);
    u.h[1] = *(const v8bf*)(base + 16 + 8 * half);
    return u.v;
}

__device__ __forceinline__ v8f wmma_bf16(v16bf a, v16bf b, v8f c) {
    return __builtin_amdgcn_wmma_f32_16x16x32_bf16(
        false, a, false, b, (short)0, c, false, false);
}

// Async memory->LDS copy of 16 bytes (no VGPR data path, tracked by ASYNCcnt).
// dst: LDS byte offset (low 32 bits of the flat LDS-aperture address),
// src: 64-bit global address. ISA 10. / 15.18.3 GLOBAL_LOAD_ASYNC_TO_LDS_B128.
__device__ __forceinline__ void async_copy_b128(void* lds_dst, const void* gsrc) {
    unsigned int       dst = (unsigned int)(unsigned long long)lds_dst;
    unsigned long long src = (unsigned long long)gsrc;
    asm volatile("global_load_async_to_lds_b128 %0, %1, off"
                 :: "v"(dst), "v"(src) : "memory");
}
__device__ __forceinline__ void wait_asynccnt0() {
    asm volatile("s_wait_asynccnt 0" ::: "memory");
}

// ---------------------------------------------------------------------------
// GEMM: C[M,N] = (A[M,K] @ W[N,K]^T + bias[N]) * scale
//   A_F32:  A is fp32 (converted to bf16 while staging); else A is bf16 and is
//           staged with async global->LDS copies.
//   OUT_F32: C stored fp32; else bf16.
// Block: 256 threads (8 waves), tile 128(M) x 64(N), K-step 32.
// Waves laid out 4(M) x 2(N); each wave computes 2x2 16x16 WMMA tiles.
// ---------------------------------------------------------------------------
template <bool A_F32, bool OUT_F32>
__global__ __launch_bounds__(256) void gemm_xwt(
    const void* __restrict__ Ap, const float* __restrict__ W,
    const float* __restrict__ bias, void* __restrict__ Cp,
    int M, int N, int K, float scale)
{
    __shared__ __bf16 As[128][40];   // 32 k + 8 pad
    __shared__ __bf16 Bs[64][40];

    const int tid  = threadIdx.x;
    const int wid  = tid >> 5;
    const int lane = tid & 31;
    const int half = lane >> 4;
    const int lm   = lane & 15;
    const int waveM = wid & 3;       // 0..3 -> 32-row slice
    const int waveN = wid >> 2;      // 0..1 -> 32-col slice
    const int bm = blockIdx.y * 128;
    const int bn = blockIdx.x * 64;

    v8f acc[2][2];
    for (int i = 0; i < 2; ++i)
        for (int j = 0; j < 2; ++j)
            acc[i][j] = {};

    for (int k0 = 0; k0 < K; k0 += 32) {
        __syncthreads();
        // ---- stage A tile (128 x 32) ----
        if (A_F32) {
            const float* A = (const float*)Ap;
            for (int it = 0; it < 4; ++it) {
                int idx = it * 256 + tid;        // 1024 float4 chunks
                int row = idx >> 3;
                int c4  = idx & 7;
                const float* src = A + (size_t)(bm + row) * K + k0 + 4 * c4;
                float4 va = *(const float4*)src;
                // Prefetch next K-window of this row (HBM-resident activations).
                if (c4 == 0 && k0 + 32 < K) __builtin_prefetch(src + 32, 0, 1);
                __bf16* d = &As[row][4 * c4];
                d[0] = (__bf16)va.x; d[1] = (__bf16)va.y;
                d[2] = (__bf16)va.z; d[3] = (__bf16)va.w;
            }
        } else {
            const __bf16* A = (const __bf16*)Ap;
            for (int it = 0; it < 2; ++it) {
                int idx = it * 256 + tid;        // 512 16-byte chunks
                int row = idx >> 2;
                int c8  = idx & 3;
                async_copy_b128(&As[row][8 * c8],
                                A + (size_t)(bm + row) * K + k0 + 8 * c8);
            }
        }
        // ---- stage B tile (64 x 32) from fp32 W[N,K] (L2-resident) ----
        for (int it = 0; it < 2; ++it) {
            int idx = it * 256 + tid;            // 512 float4 chunks
            int row = idx >> 3;
            int c4  = idx & 7;
            float4 vb = *(const float4*)(W + (size_t)(bn + row) * K + k0 + 4 * c4);
            __bf16* d = &Bs[row][4 * c4];
            d[0] = (__bf16)vb.x; d[1] = (__bf16)vb.y;
            d[2] = (__bf16)vb.z; d[3] = (__bf16)vb.w;
        }
        if (!A_F32) wait_asynccnt0();
        __syncthreads();

        v16bf af[2], bvf[2];
        af[0]  = load_a_frag(&As[32 * waveM + lm][0], half);
        af[1]  = load_a_frag(&As[32 * waveM + 16 + lm][0], half);
        bvf[0] = load_b_frag(&Bs[32 * waveN + lm][0], half);
        bvf[1] = load_b_frag(&Bs[32 * waveN + 16 + lm][0], half);
        for (int i = 0; i < 2; ++i)
            for (int j = 0; j < 2; ++j)
                acc[i][j] = wmma_bf16(af[i], bvf[j], acc[i][j]);
    }

    // ---- epilogue: C row = r + 8*half, col = lm (per 16x16 tile) ----
    for (int i = 0; i < 2; ++i) {
        for (int j = 0; j < 2; ++j) {
            int col = bn + 32 * waveN + 16 * j + lm;
            float bv = bias[col];
            for (int r = 0; r < 8; ++r) {
                int row = bm + 32 * waveM + 16 * i + r + 8 * half;
                float v = (acc[i][j][r] + bv) * scale;
                if (OUT_F32)
                    ((float*)Cp)[(size_t)row * N + col] = v;
                else
                    ((__bf16*)Cp)[(size_t)row * N + col] = (__bf16)v;
            }
        }
    }
}

// ---------------------------------------------------------------------------
// Flash attention: grid (S/64, H, B); block 128 (4 waves, 16 q-rows each).
// Q already scaled by 1/sqrt(D). Streams 64-key tiles; online softmax.
// K tile staged with async global->LDS; V staged transposed via VALU.
// ---------------------------------------------------------------------------
__global__ __launch_bounds__(128) void attn_kernel(
    const __bf16* __restrict__ Qb, const __bf16* __restrict__ Kb,
    const __bf16* __restrict__ Vb, __bf16* __restrict__ Cx)
{
    __shared__ __bf16 Kt[64][72];        // [key][d]
    __shared__ __bf16 Vt[64][72];        // transposed: [d][key]
    __shared__ __bf16 Pb[4][16][72];     // per-wave P tile [qrow][key]

    const int qt  = blockIdx.x;
    const int h   = blockIdx.y;
    const int b   = blockIdx.z;
    const int tid = threadIdx.x;
    const int wid = tid >> 5;
    const int lane = tid & 31;
    const int half = lane >> 4;
    const int lm   = lane & 15;

    // Load this wave's Q fragments once (A-matrix layout, D=64 -> 2 k-steps).
    const __bf16* qrow_ptr =
        Qb + (size_t)(b * ATTN_S + qt * 64 + wid * 16 + lm) * ATTN_SZ + h * ATTN_D;
    v16bf qf[2];
    for (int s = 0; s < 2; ++s) {
        FragU u;
        u.h[0] = *(const v8bf*)(qrow_ptr + 32 * s + 8 * half);
        u.h[1] = *(const v8bf*)(qrow_ptr + 32 * s + 16 + 8 * half);
        qf[s] = u.v;
    }

    float mrun[8], lrun[8];
    v8f cacc[4];
    for (int r = 0; r < 8; ++r) { mrun[r] = -1e30f; lrun[r] = 0.0f; }
    for (int j = 0; j < 4; ++j) cacc[j] = {};

    for (int kt2 = 0; kt2 < ATTN_S / 64; ++kt2) {
        __syncthreads();
        // ---- stage K (async, row-major) and V (transposed, VALU) tiles ----
        const int krow0 = kt2 * 64;
        for (int it = 0; it < 4; ++it) {
            int idx = it * 128 + tid;            // 512 16-byte chunks
            int row = idx >> 3;                  // key 0..63
            int c8  = idx & 7;                   // d-chunk
            size_t gro = (size_t)(b * ATTN_S + krow0 + row) * ATTN_SZ + h * ATTN_D + 8 * c8;
            async_copy_b128(&Kt[row][8 * c8], Kb + gro);
            v8bf vv = *(const v8bf*)(Vb + gro);
            for (int e = 0; e < 8; ++e) Vt[8 * c8 + e][row] = vv[e];
        }
        wait_asynccnt0();
        __syncthreads();

        // ---- scores S = Q @ K^T : 4 n-tiles x 2 k-steps
        v8f sacc[4];
        for (int j = 0; j < 4; ++j) sacc[j] = {};
        for (int j = 0; j < 4; ++j)
            for (int s = 0; s < 2; ++s)
                sacc[j] = wmma_bf16(qf[s], load_b_frag(&Kt[16 * j + lm][32 * s], half), sacc[j]);

        // ---- online softmax (rows r+8*half; row entries striped on 16 lanes)
        float pr[4][8];
        for (int r = 0; r < 8; ++r) {
            float v = fmaxf(fmaxf(sacc[0][r], sacc[1][r]),
                            fmaxf(sacc[2][r], sacc[3][r]));
            v = fmaxf(v, __shfl_xor(v, 1, 32));
            v = fmaxf(v, __shfl_xor(v, 2, 32));
            v = fmaxf(v, __shfl_xor(v, 4, 32));
            v = fmaxf(v, __shfl_xor(v, 8, 32));
            float mn = fmaxf(mrun[r], v);
            float sc = __expf(mrun[r] - mn);
            mrun[r] = mn;
            float sum = 0.0f;
            for (int j = 0; j < 4; ++j) {
                float p = __expf(sacc[j][r] - mn);
                pr[j][r] = p;
                sum += p;
            }
            sum += __shfl_xor(sum, 1, 32);
            sum += __shfl_xor(sum, 2, 32);
            sum += __shfl_xor(sum, 4, 32);
            sum += __shfl_xor(sum, 8, 32);
            lrun[r] = lrun[r] * sc + sum;
            for (int j = 0; j < 4; ++j) cacc[j][r] *= sc;
        }

        // ---- C-layout P -> A-layout via per-wave LDS tile
        for (int j = 0; j < 4; ++j)
            for (int r = 0; r < 8; ++r)
                Pb[wid][r + 8 * half][16 * j + lm] = (__bf16)pr[j][r];

        // ---- ctx += P @ V : 4 d-tiles x 2 k-steps over keys
        for (int j = 0; j < 4; ++j)
            for (int s = 0; s < 2; ++s) {
                v16bf pa = load_a_frag(&Pb[wid][lm][32 * s], half);
                v16bf vb = load_b_frag(&Vt[16 * j + lm][32 * s], half);
                cacc[j] = wmma_bf16(pa, vb, cacc[j]);
            }
    }

    // ---- normalize and store merged-head ctx (bf16)
    float inv[8];
    for (int r = 0; r < 8; ++r) inv[r] = 1.0f / lrun[r];
    for (int j = 0; j < 4; ++j)
        for (int r = 0; r < 8; ++r) {
            size_t row = (size_t)(b * ATTN_S + qt * 64 + wid * 16 + r + 8 * half);
            Cx[row * ATTN_SZ + h * ATTN_D + 16 * j + lm] = (__bf16)(cacc[j][r] * inv[r]);
        }
}

// ---------------------------------------------------------------------------
extern "C" void kernel_launch(void* const* d_in, const int* in_sizes, int n_in,
                              void* d_out, int out_size, void* d_ws, size_t ws_size,
                              hipStream_t stream) {
    const float* q  = (const float*)d_in[0];
    const float* k  = (const float*)d_in[1];
    const float* v  = (const float*)d_in[2];
    const float* Wq = (const float*)d_in[3];
    const float* bq = (const float*)d_in[4];
    const float* Wk = (const float*)d_in[5];
    const float* bk = (const float*)d_in[6];
    const float* Wv = (const float*)d_in[7];
    const float* bv = (const float*)d_in[8];
    const float* Wo = (const float*)d_in[9];
    const float* bo = (const float*)d_in[10];

    const int M = ATTN_B * ATTN_S;            // 4096
    const int N = ATTN_SZ;                    // 1024
    const int K = ATTN_SZ;                    // 1024
    const size_t MAT = (size_t)M * ATTN_SZ;   // 4096*1024 elems

    __bf16* Qb = (__bf16*)d_ws;               // needs 4*MAT*2 = 32 MB workspace
    __bf16* Kb = Qb + MAT;
    __bf16* Vb = Kb + MAT;
    __bf16* Cx = Vb + MAT;

    dim3 gg(N / 64, M / 128);                 // (16, 32)

    // Projections (Q carries the 1/sqrt(D) scale).
    gemm_xwt<true,  false><<<gg, 256, 0, stream>>>(q, Wq, bq, Qb, M, N, K, 0.125f);
    gemm_xwt<true,  false><<<gg, 256, 0, stream>>>(k, Wk, bk, Kb, M, N, K, 1.0f);
    gemm_xwt<true,  false><<<gg, 256, 0, stream>>>(v, Wv, bv, Vb, M, N, K, 1.0f);

    // Attention: grid (qtiles, heads, batch).
    attn_kernel<<<dim3(ATTN_S / 64, ATTN_H, ATTN_B), 128, 0, stream>>>(Qb, Kb, Vb, Cx);

    // Output projection to fp32.
    gemm_xwt<false, true><<<gg, 256, 0, stream>>>(Cx, Wo, bo, (float*)d_out, M, N, K, 1.0f);
}
